// DTIProfilerCrossAttention_55671366090757
// MI455X (gfx1250) — compile-verified, matrix operand order checked by reading, not run on hardware
//
#include <hip/hip_runtime.h>

// ---------------------------------------------------------------------------
// DTI model with len-1 sequences: softmax over a single key == 1, so every
// attention block reduces to Wo(Wv x + bv) + bo (q/k projections are dead).
// Whole model = GEMM chain + bias/relu/residual-LayerNorm + sigmoid head.
// ~374 GFLOP vs ~235MB traffic -> compute bound; run GEMMs on
// v_wmma_f32_16x16x32_f16 (f16 in, f32 accumulate). Wave tile 32x64
// (block tile 128x128): 8 WMMA per 6 fragment loads to keep VMEM issue
// off the critical path; weights stay L2-resident (11.5MB << 192MB L2),
// A stream prefetched ahead via global_prefetch_b8.
// ---------------------------------------------------------------------------

typedef _Float16 h16;
typedef __attribute__((ext_vector_type(16))) _Float16 v16h;
typedef __attribute__((ext_vector_type(8)))  _Float16 v8h;
typedef __attribute__((ext_vector_type(4)))  _Float16 v4h;
typedef __attribute__((ext_vector_type(8)))  float    v8f;

union FragU { v16h v; struct { v8h lo, hi; } s; };

// CDNA5 16-bit A/B fragment (16 rows x 32 K) from K-contiguous rows.
// lanes 0-15: row=lane, VGPR0-3 K=0..7,  VGPR4-7 K=16..23
// lanes16-31: row=lane-16, VGPR0-3 K=8..15, VGPR4-7 K=24..31
__device__ __forceinline__ v16h load_frag(const h16* __restrict__ base, int ld) {
  const int lane = threadIdx.x & 31;
  const h16* p = base + (size_t)(lane & 15) * ld + (lane >> 4) * 8;
  FragU u;
  u.s.lo = *(const v8h*)(p);        // K = lh*8 + 0..7
  u.s.hi = *(const v8h*)(p + 16);   // K = lh*8 + 16..23
  return u.v;
}

// C = act(A @ W^T + bias), A: MxK row-major f16, W: NxK row-major f16.
// Workgroup tile 128x128, 8 waves (4Mx2N), each wave 32x64 = 8 accumulators.
template <int ACT, int OUTF, int OUTH>
__global__ __launch_bounds__(256)
void gemm_f16_kernel(const h16* __restrict__ A, const h16* __restrict__ W,
                     const float* __restrict__ bias,
                     float* __restrict__ Cf, h16* __restrict__ Ch,
                     int M, int N, int K) {
  const int lane = threadIdx.x & 31;
  const int wave = threadIdx.x >> 5;
  const int m0 = blockIdx.x * 128 + (wave >> 1) * 32;
  const int n0 = blockIdx.y * 128 + (wave & 1) * 64;

  const h16* Ap0 = A + (size_t)m0 * K;
  const h16* Ap1 = Ap0 + (size_t)16 * K;
  const h16* Wp0 = W + (size_t)n0 * K;
  const h16* Wp1 = Wp0 + (size_t)16 * K;
  const h16* Wp2 = Wp0 + (size_t)32 * K;
  const h16* Wp3 = Wp0 + (size_t)48 * K;

  v8f acc[2][4] = {};
  for (int k = 0; k < K; k += 32) {
    // Pull the HBM-resident A stream ahead (speculative; OOB dropped).
    __builtin_prefetch((const void*)(Ap0 + k + 64), 0, 0);
    __builtin_prefetch((const void*)(Ap1 + k + 64), 0, 0);
    v16h a0 = load_frag(Ap0 + k, K);
    v16h a1 = load_frag(Ap1 + k, K);
    v16h b0 = load_frag(Wp0 + k, K);
    v16h b1 = load_frag(Wp1 + k, K);
    v16h b2 = load_frag(Wp2 + k, K);
    v16h b3 = load_frag(Wp3 + k, K);
    acc[0][0] = __builtin_amdgcn_wmma_f32_16x16x32_f16(false, a0, false, b0, (short)0, acc[0][0], false, false);
    acc[0][1] = __builtin_amdgcn_wmma_f32_16x16x32_f16(false, a0, false, b1, (short)0, acc[0][1], false, false);
    acc[0][2] = __builtin_amdgcn_wmma_f32_16x16x32_f16(false, a0, false, b2, (short)0, acc[0][2], false, false);
    acc[0][3] = __builtin_amdgcn_wmma_f32_16x16x32_f16(false, a0, false, b3, (short)0, acc[0][3], false, false);
    acc[1][0] = __builtin_amdgcn_wmma_f32_16x16x32_f16(false, a1, false, b0, (short)0, acc[1][0], false, false);
    acc[1][1] = __builtin_amdgcn_wmma_f32_16x16x32_f16(false, a1, false, b1, (short)0, acc[1][1], false, false);
    acc[1][2] = __builtin_amdgcn_wmma_f32_16x16x32_f16(false, a1, false, b2, (short)0, acc[1][2], false, false);
    acc[1][3] = __builtin_amdgcn_wmma_f32_16x16x32_f16(false, a1, false, b3, (short)0, acc[1][3], false, false);
  }

  // 16x16 f32 C layout: VGPR r -> M=r (lanes 0-15), M=8+r (lanes 16-31); N=lane&15
  const int col = lane & 15;
  const int rb  = (lane >> 4) * 8;
#pragma unroll
  for (int mi = 0; mi < 2; ++mi) {
#pragma unroll
    for (int nj = 0; nj < 4; ++nj) {
      const int gcol = n0 + nj * 16 + col;
      const float bv = bias[gcol];
#pragma unroll
      for (int r = 0; r < 8; ++r) {
        const int grow = m0 + mi * 16 + rb + r;
        float v = acc[mi][nj][r] + bv;
        if (ACT) v = v > 0.f ? v : 0.f;
        const size_t off = (size_t)grow * N + gcol;
        if (OUTF) Cf[off] = v;
        if (OUTH) Ch[off] = (h16)v;
      }
    }
  }
  (void)M;
}

// out = LayerNorm(x + y) over D=512, one wave per row (16 elems/lane).
__global__ __launch_bounds__(256)
void ln_add_kernel(const float* __restrict__ x, const float* __restrict__ y,
                   const float* __restrict__ g, const float* __restrict__ b,
                   float* __restrict__ outf, h16* __restrict__ outh,
                   int ldh, int M) {
  const int lane = threadIdx.x & 31;
  const int row  = blockIdx.x * 8 + (threadIdx.x >> 5);
  if (row >= M) return;
  const float* xr = x + (size_t)row * 512;
  const float* yr = y + (size_t)row * 512;
  float vals[16];
  float s = 0.f;
#pragma unroll
  for (int i = 0; i < 16; ++i) {
    const int c = lane * 16 + i;
    vals[i] = xr[c] + yr[c];
    s += vals[i];
  }
#pragma unroll
  for (int off = 16; off > 0; off >>= 1) s += __shfl_xor(s, off, 32);
  const float mean = s * (1.f / 512.f);
  float vs = 0.f;
#pragma unroll
  for (int i = 0; i < 16; ++i) { const float d = vals[i] - mean; vs += d * d; }
#pragma unroll
  for (int off = 16; off > 0; off >>= 1) vs += __shfl_xor(vs, off, 32);
  const float rstd = rsqrtf(vs * (1.f / 512.f) + 1e-5f);
#pragma unroll
  for (int i = 0; i < 16; ++i) {
    const int c = lane * 16 + i;
    const float o = (vals[i] - mean) * rstd * g[c] + b[c];
    if (outf) outf[(size_t)row * 512 + c] = o;
    if (outh) outh[(size_t)row * ldh + c] = (h16)o;
  }
}

// f32 -> f16 conversion, 4 elems/thread (all sizes are multiples of 4).
__global__ __launch_bounds__(256)
void cvt4_kernel(const float* __restrict__ s, h16* __restrict__ d, long n4) {
  const long i = (long)blockIdx.x * 256 + threadIdx.x;
  if (i >= n4) return;
  const float4 f = ((const float4*)s)[i];
  v4h h;
  h[0] = (h16)f.x; h[1] = (h16)f.y; h[2] = (h16)f.z; h[3] = (h16)f.w;
  ((v4h*)d)[i] = h;
}

// out[row] = sigmoid(dot(h[row, 0:256], w) + b), one wave per row.
__global__ __launch_bounds__(256)
void head_kernel(const h16* __restrict__ h, const float* __restrict__ w,
                 const float* __restrict__ b, float* __restrict__ out, int M) {
  const int lane = threadIdx.x & 31;
  const int row  = blockIdx.x * 8 + (threadIdx.x >> 5);
  if (row >= M) return;
  const h16* hr = h + (size_t)row * 256;
  float s = 0.f;
#pragma unroll
  for (int i = 0; i < 8; ++i) {
    const int c = lane * 8 + i;
    s += (float)hr[c] * w[c];
  }
#pragma unroll
  for (int off = 16; off > 0; off >>= 1) s += __shfl_xor(s, off, 32);
  if (lane == 0) out[row] = 1.f / (1.f + __expf(-(s + b[0])));
}

// ---------------------------------------------------------------------------

static inline void launch_gemm(int act, int outf, int outh,
                               const h16* A, const h16* W, const float* bias,
                               float* Cf, h16* Ch, int M, int N, int K,
                               hipStream_t s) {
  dim3 g(M / 128, N / 128), b(256);
  if (act == 0 && outf && outh)
    gemm_f16_kernel<0, 1, 1><<<g, b, 0, s>>>(A, W, bias, Cf, Ch, M, N, K);
  else if (act == 0 && !outf && outh)
    gemm_f16_kernel<0, 0, 1><<<g, b, 0, s>>>(A, W, bias, Cf, Ch, M, N, K);
  else if (act == 0 && outf && !outh)
    gemm_f16_kernel<0, 1, 0><<<g, b, 0, s>>>(A, W, bias, Cf, Ch, M, N, K);
  else
    gemm_f16_kernel<1, 0, 1><<<g, b, 0, s>>>(A, W, bias, Cf, Ch, M, N, K);
}

static inline void launch_cvt(const float* s, h16* d, size_t n, hipStream_t st) {
  const long n4 = (long)(n / 4);
  cvt4_kernel<<<dim3((unsigned)((n4 + 255) / 256)), dim3(256), 0, st>>>(s, d, n4);
}

extern "C" void kernel_launch(void* const* d_in, const int* in_sizes, int n_in,
                              void* d_out, int out_size, void* d_ws, size_t ws_size,
                              hipStream_t stream) {
  (void)n_in; (void)ws_size; (void)out_size;
  const int B = in_sizes[0] / 768;  // 32768, multiple of 128

  const float* drug   = (const float*)d_in[0];
  const float* target = (const float*)d_in[1];
  auto F = [&](int i) { return (const float*)d_in[i]; };
  // d_in index map (setup_inputs dict insertion order, params flattened):
  //  2 prot_proj_w  3 prot_proj_b  4 mol_proj_w  5 mol_proj_b
  //  prot_enc: 6 wq 7 bq 8 wk 9 bk 10 wv 11 bv 12 wo 13 bo
  //            14 ff1_w 15 ff1_b 16 ff2_w 17 ff2_b 18 ln1_g 19 ln1_b 20 ln2_g 21 ln2_b
  //  mol_enc:  22..37 (same layout)
  //  p2m: 38 wq 39 bq 40 wk 41 bk 42 wv 43 bv 44 wo 45 bo 46 ln_g 47 ln_b
  //  m2p: 48..57 (same layout)
  //  58 c1_w 59 c1_b 60 c2_w 61 c2_b 62 c3_w 63 c3_b

  // f16 weight pool offsets (element offsets).
  size_t o = 0;
  const size_t W_PROT = o; o += 512 * 1024;
  const size_t W_MOL  = o; o += 512 * 768;
  const size_t PE_WV  = o; o += 512 * 512;
  const size_t PE_WO  = o; o += 512 * 512;
  const size_t PE_F1  = o; o += 1024 * 512;
  const size_t PE_F2  = o; o += 512 * 1024;
  const size_t ME_WV  = o; o += 512 * 512;
  const size_t ME_WO  = o; o += 512 * 512;
  const size_t ME_F1  = o; o += 1024 * 512;
  const size_t ME_F2  = o; o += 512 * 1024;
  const size_t XP_WV  = o; o += 512 * 512;   // p2m
  const size_t XP_WO  = o; o += 512 * 512;
  const size_t XM_WV  = o; o += 512 * 512;   // m2p
  const size_t XM_WO  = o; o += 512 * 512;
  const size_t C1W    = o; o += 512 * 1024;
  const size_t C2W    = o; o += 256 * 512;
  const size_t WTOT   = o;

  // Workspace carve (~364 MB for B=32768).
  char* ws = (char*)d_ws;
  auto carve = [&](size_t bytes) {
    char* p = ws;
    ws += (bytes + 255) & ~(size_t)255;
    return p;
  };
  h16*   Wh      = (h16*)  carve(sizeof(h16) * WTOT);
  float* xprot_f = (float*)carve(4ull * B * 512);
  float* xmol_f  = (float*)carve(4ull * B * 512);
  float* tmp_f   = (float*)carve(4ull * B * 512);
  h16*   xprot_h = (h16*)  carve(2ull * B * 512);
  h16*   xmol_h  = (h16*)  carve(2ull * B * 512);
  h16*   v_h     = (h16*)  carve(2ull * B * 512);
  h16*   big_h   = (h16*)  carve(2ull * B * 1024); // target/drug f16, FF hidden, feat

  // 1) Convert weights to f16 (every call; deterministic, no caching).
  struct { int idx; size_t off, n; } wc[] = {
      {2, W_PROT, 512ull * 1024}, {4, W_MOL, 512ull * 768},
      {10, PE_WV, 512ull * 512}, {12, PE_WO, 512ull * 512},
      {14, PE_F1, 1024ull * 512}, {16, PE_F2, 512ull * 1024},
      {26, ME_WV, 512ull * 512}, {28, ME_WO, 512ull * 512},
      {30, ME_F1, 1024ull * 512}, {32, ME_F2, 512ull * 1024},
      {42, XP_WV, 512ull * 512}, {44, XP_WO, 512ull * 512},
      {52, XM_WV, 512ull * 512}, {54, XM_WO, 512ull * 512},
      {58, C1W, 512ull * 1024}, {60, C2W, 256ull * 512},
  };
  for (auto& w : wc) launch_cvt(F(w.idx), Wh + w.off, w.n, stream);

  const dim3 lnGrid(B / 8), lnBlk(256);
  auto ln = [&](const float* x, const float* y, const float* g, const float* bb,
                float* of, h16* oh, int ldh) {
    ln_add_kernel<<<lnGrid, lnBlk, 0, stream>>>(x, y, g, bb, of, oh, ldh, B);
  };

  // 2) Input projections.
  launch_cvt(target, big_h, (size_t)B * 1024, stream);
  launch_gemm(0, 1, 1, big_h, Wh + W_PROT, F(3), xprot_f, xprot_h, B, 512, 1024, stream);
  launch_cvt(drug, big_h, (size_t)B * 768, stream);
  launch_gemm(0, 1, 1, big_h, Wh + W_MOL, F(5), xmol_f, xmol_h, B, 512, 768, stream);

  // 3) Encoder layers (attention == Wo(Wv x + bv) + bo for len-1 seq).
  auto encoder = [&](float* x_f, h16* x_h, size_t wv, int bv, size_t wo, int bo,
                     size_t f1, int b1, size_t f2, int b2,
                     int l1g, int l1b, int l2g, int l2b) {
    launch_gemm(0, 0, 1, x_h, Wh + wv, F(bv), nullptr, v_h, B, 512, 512, stream);
    launch_gemm(0, 1, 0, v_h, Wh + wo, F(bo), tmp_f, nullptr, B, 512, 512, stream);
    ln(x_f, tmp_f, F(l1g), F(l1b), x_f, x_h, 512);
    launch_gemm(1, 0, 1, x_h, Wh + f1, F(b1), nullptr, big_h, B, 1024, 512, stream);
    launch_gemm(0, 1, 0, big_h, Wh + f2, F(b2), tmp_f, nullptr, B, 512, 1024, stream);
    ln(x_f, tmp_f, F(l2g), F(l2b), x_f, x_h, 512);
  };
  encoder(xprot_f, xprot_h, PE_WV, 11, PE_WO, 13, PE_F1, 15, PE_F2, 17, 18, 19, 20, 21);
  encoder(xmol_f,  xmol_h,  ME_WV, 27, ME_WO, 29, ME_F1, 31, ME_F2, 33, 34, 35, 36, 37);

  // 4) Cross attention; LN results go straight into feat (B x 1024, f16) = big_h.
  // p2m: ctx from mol, residual prot -> feat[:, 0:512]
  launch_gemm(0, 0, 1, xmol_h, Wh + XP_WV, F(43), nullptr, v_h, B, 512, 512, stream);
  launch_gemm(0, 1, 0, v_h, Wh + XP_WO, F(45), tmp_f, nullptr, B, 512, 512, stream);
  ln(xprot_f, tmp_f, F(46), F(47), nullptr, big_h, 1024);
  // m2p: ctx from prot, residual mol -> feat[:, 512:1024]
  launch_gemm(0, 0, 1, xprot_h, Wh + XM_WV, F(53), nullptr, v_h, B, 512, 512, stream);
  launch_gemm(0, 1, 0, v_h, Wh + XM_WO, F(55), tmp_f, nullptr, B, 512, 512, stream);
  ln(xmol_f, tmp_f, F(56), F(57), nullptr, big_h + 512, 1024);

  // 5) Classifier.
  launch_gemm(1, 0, 1, big_h, Wh + C1W, F(59), nullptr, v_h, B, 512, 1024, stream);
  h16* h2 = xprot_h;  // dead by now; B x 256 fits
  launch_gemm(1, 0, 1, v_h, Wh + C2W, F(61), nullptr, h2, B, 256, 512, stream);
  head_kernel<<<dim3(B / 8), dim3(256), 0, stream>>>(h2, F(62), F(63), (float*)d_out, B);
}
